// GraphSAGEGraphLevel_71674414235947
// MI455X (gfx1250) — matrix operand order repaired and checked by reading, not compiled, not run on hardware
//
#include <hip/hip_runtime.h>

// ---------------------------------------------------------------------------
// Types for WMMA operands (CDNA5 / gfx1250, wave32)
// ---------------------------------------------------------------------------
typedef __attribute__((ext_vector_type(16))) __bf16 v16bf;
typedef __attribute__((ext_vector_type(8)))  __bf16 v8bf;
typedef __attribute__((ext_vector_type(8)))  float  v8f;

#define N_NODES 50000
#define N_EDGES 800000
#define IN_DIM  116
#define EMB_DIM 12
#define HID     128
#define OUT_DIM 10
#define NUM_GRAPHS 64

// ---------------------------------------------------------------------------
// bf16 helpers (round-to-nearest-even), no header dependency
// ---------------------------------------------------------------------------
static __device__ __forceinline__ __bf16 f2bf(float f) {
  unsigned u = __builtin_bit_cast(unsigned, f);
  unsigned r = u + 0x7FFFu + ((u >> 16) & 1u);
  unsigned short s = (unsigned short)(r >> 16);
  return __builtin_bit_cast(__bf16, s);
}
static __device__ __forceinline__ float bf2f(__bf16 b) {
  unsigned short s = __builtin_bit_cast(unsigned short, b);
  unsigned u = ((unsigned)s) << 16;
  return __builtin_bit_cast(float, u);
}

// ---------------------------------------------------------------------------
// Utility kernels
// ---------------------------------------------------------------------------
__global__ void k_zero(float* __restrict__ p, long long n) {
  long long t = (long long)blockIdx.x * blockDim.x + threadIdx.x;
  if (t < n) p[t] = 0.0f;
}

__global__ void k_degree(const int* __restrict__ edst, float* __restrict__ cnt, int nE) {
  int e = blockIdx.x * blockDim.x + threadIdx.x;
  if (e < nE) atomicAdd(&cnt[edst[e]], 1.0f);
}

__global__ void k_rcnt(float* __restrict__ cnt, int n) {
  int i = blockIdx.x * blockDim.x + threadIdx.x;
  if (i < n) cnt[i] = 1.0f / fmaxf(cnt[i], 1.0f);
}

// h0 = concat([x, emb[st]]) in bf16; feature dim = 116 + 12 = 128
__global__ void k_build_h0(const float* __restrict__ x, const int* __restrict__ st,
                           const float* __restrict__ emb, __bf16* __restrict__ h0,
                           int nNodes) {
  long long t = (long long)blockIdx.x * blockDim.x + threadIdx.x;
  if (t >= (long long)nNodes * HID) return;
  int i = (int)(t >> 7);
  int k = (int)(t & 127);
  float v = (k < IN_DIM) ? x[(long long)i * IN_DIM + k]
                         : emb[(long long)st[i] * EMB_DIM + (k - IN_DIM)];
  h0[t] = f2bf(v);
}

__global__ void k_f32_to_bf16(const float* __restrict__ s, __bf16* __restrict__ d, long long n) {
  long long t = (long long)blockIdx.x * blockDim.x + threadIdx.x;
  if (t < n) d[t] = f2bf(s[t]);
}

// agg[dst] += h[src]  (one thread per (edge, feature); coalesced across k)
__global__ void k_scatter(const int* __restrict__ esrc, const int* __restrict__ edst,
                          const __bf16* __restrict__ h, float* __restrict__ agg, int nE) {
  long long t = (long long)blockIdx.x * blockDim.x + threadIdx.x;
  if (t >= (long long)nE * HID) return;
  int e = (int)(t >> 7);
  int k = (int)(t & 127);
  int s = esrc[e], d = edst[e];
  float v = bf2f(h[(long long)s * HID + k]);
  atomicAdd(&agg[(long long)d * HID + k], v);
}

// meanbf = bf16(agg * rcnt[node])
__global__ void k_meanbf(const float* __restrict__ agg, const float* __restrict__ rcnt,
                         __bf16* __restrict__ mbf, long long n) {
  long long t = (long long)blockIdx.x * blockDim.x + threadIdx.x;
  if (t >= n) return;
  int i = (int)(t >> 7);
  mbf[t] = f2bf(agg[t] * rcnt[i]);
}

// ---------------------------------------------------------------------------
// Fused SAGE layer GEMM:  out = relu(A1 @ W1^T + bias + A2 @ W2^T)
// A1,A2: [M x 128] bf16 row-major; W1,W2: [128 x 128] bf16 row-major (fanout,fanin)
// One wave per 16-row tile; 8 column tiles; v_wmma_f32_16x16x32_bf16.
// ---------------------------------------------------------------------------
__global__ __launch_bounds__(256)
void k_gemm_sage(const __bf16* __restrict__ A1, const __bf16* __restrict__ W1,
                 const float* __restrict__ bias,
                 const __bf16* __restrict__ A2, const __bf16* __restrict__ W2,
                 float* __restrict__ outF, __bf16* __restrict__ outB, int M) {
  const int K = HID, N = HID;
  const int wave = threadIdx.x >> 5;
  const int lane = threadIdx.x & 31;
  const int mt = blockIdx.x * 8 + wave;
  const int m0 = mt * 16;
  if (m0 >= M) return;                 // wave-uniform: EXEC stays all-1s inside
  const int lhalf = lane >> 4;         // 0 | 1
  const int l16 = lane & 15;

  v8f acc[8];
#pragma unroll
  for (int i = 0; i < 8; ++i)
#pragma unroll
    for (int j = 0; j < 8; ++j) acc[i][j] = 0.0f;

#pragma unroll
  for (int pass = 0; pass < 2; ++pass) {
    const __bf16* A = pass ? A2 : A1;
    const __bf16* W = pass ? W2 : W1;
    const __bf16* arow = A + (long long)(m0 + l16) * K;
#pragma unroll
    for (int kc = 0; kc < 4; ++kc) {
      const int k0 = kc * 32;
      // A fragment (contiguous k order): k 0..15 (lanes 0-15) / 16..31 (lanes 16-31)
      const int ka = k0 + lhalf * 16;
      v8bf alo = *(const v8bf*)(arow + ka);
      v8bf ahi = *(const v8bf*)(arow + ka + 8);
      v16bf a = __builtin_shufflevector(alo, ahi,
                  0,1,2,3,4,5,6,7,8,9,10,11,12,13,14,15);
      // B fragment: lane l<16 -> k 0..15 (same order as A) = W[n][k0..k0+15]; lanes 16-31 -> k0+16..k0+31
      const int kb = k0 + lhalf * 16;
#pragma unroll
      for (int nt = 0; nt < 8; ++nt) {
        const __bf16* wrow = W + (long long)(nt * 16 + l16) * K + kb;
        v16bf b = *(const v16bf*)wrow;
        acc[nt] = __builtin_amdgcn_wmma_f32_16x16x32_bf16(
            /*neg_a=*/false, a, /*neg_b=*/false, b,
            /*c_mod=*/(short)0, acc[nt], /*reuse_a=*/false, /*reuse_b=*/false);
      }
    }
  }

  // D layout: VGPR r, lanes 0-15 -> (M=r, N=lane); lanes 16-31 -> (M=8+r, N=lane-16)
#pragma unroll
  for (int nt = 0; nt < 8; ++nt) {
    const int n = nt * 16 + l16;
    const float bv = bias[n];
#pragma unroll
    for (int r = 0; r < 8; ++r) {
      const int m = m0 + r + 8 * lhalf;
      float v = acc[nt][r] + bv;
      v = v > 0.0f ? v : 0.0f;
      if (outF) outF[(long long)m * N + n] = v;
      if (outB) outB[(long long)m * N + n] = f2bf(v);
    }
  }
}

// ---------------------------------------------------------------------------
// Graph mean pooling (atomics) + output head
// ---------------------------------------------------------------------------
__global__ void k_pool(const float* __restrict__ h2, const int* __restrict__ batch,
                       float* __restrict__ gsum, float* __restrict__ gcnt, int nNodes) {
  long long t = (long long)blockIdx.x * blockDim.x + threadIdx.x;
  if (t >= (long long)nNodes * HID) return;
  int i = (int)(t >> 7);
  int k = (int)(t & 127);
  int g = batch[i];
  atomicAdd(&gsum[(long long)g * HID + k], h2[t]);
  if (k == 0) atomicAdd(&gcnt[g], 1.0f);
}

__global__ void k_head(const float* __restrict__ gsum, const float* __restrict__ gcnt,
                       const float* __restrict__ Wout, const float* __restrict__ bout,
                       float* __restrict__ out) {
  int t = blockIdx.x * blockDim.x + threadIdx.x;
  if (t >= NUM_GRAPHS * OUT_DIM) return;
  int g = t / OUT_DIM;
  int o = t % OUT_DIM;
  float s = 0.0f;
  for (int k = 0; k < HID; ++k)
    s += gsum[g * HID + k] * Wout[o * HID + k];
  float r = 1.0f / fmaxf(gcnt[g], 1.0f);
  out[t] = s * r + bout[o];
}

// ---------------------------------------------------------------------------
// Launcher
// ---------------------------------------------------------------------------
extern "C" void kernel_launch(void* const* d_in, const int* in_sizes, int n_in,
                              void* d_out, int out_size, void* d_ws, size_t ws_size,
                              hipStream_t stream) {
  (void)in_sizes; (void)n_in; (void)out_size; (void)ws_size;
  const float* x    = (const float*)d_in[0];
  const int*   eidx = (const int*)d_in[1];
  /* d_in[2] edge_attr unused by reference */
  const int*   st   = (const int*)d_in[3];
  const int*   bat  = (const int*)d_in[4];
  const float* emb  = (const float*)d_in[5];
  const float* Wl1  = (const float*)d_in[6];
  const float* bl1  = (const float*)d_in[7];
  const float* Wr1  = (const float*)d_in[8];
  const float* Wl2  = (const float*)d_in[9];
  const float* bl2  = (const float*)d_in[10];
  const float* Wr2  = (const float*)d_in[11];
  const float* Wout = (const float*)d_in[12];
  const float* bout = (const float*)d_in[13];
  float* out = (float*)d_out;

  const int* esrc = eidx;
  const int* edst = eidx + N_EDGES;

  // --- workspace layout (256B aligned) ---
  char* ws = (char*)d_ws;
  size_t off = 0;
  auto alloc = [&](size_t bytes) -> char* {
    char* p = ws + off;
    off = (off + bytes + 255) & ~(size_t)255;
    return p;
  };
  const long long NF = (long long)N_NODES * HID;   // 6.4M elements
  __bf16* h0bf  = (__bf16*)alloc(NF * 2);
  __bf16* h1bf  = (__bf16*)alloc(NF * 2);
  __bf16* mbf   = (__bf16*)alloc(NF * 2);
  float*  agg   = (float*) alloc(NF * 4);
  float*  h2f   = (float*) alloc(NF * 4);
  float*  rcnt  = (float*) alloc((size_t)N_NODES * 4);
  __bf16* wl1bf = (__bf16*)alloc((size_t)HID * HID * 2);
  __bf16* wr1bf = (__bf16*)alloc((size_t)HID * HID * 2);
  __bf16* wl2bf = (__bf16*)alloc((size_t)HID * HID * 2);
  __bf16* wr2bf = (__bf16*)alloc((size_t)HID * HID * 2);
  float*  gsum  = (float*) alloc((size_t)(NUM_GRAPHS * HID + NUM_GRAPHS) * 4);
  float*  gcnt  = gsum + NUM_GRAPHS * HID;

  const int T = 256;
  const unsigned gNF  = (unsigned)((NF + T - 1) / T);                       // 25000
  const unsigned gEF  = (unsigned)(((long long)N_EDGES * HID + T - 1) / T); // 400000
  const unsigned gE   = (N_EDGES + T - 1) / T;
  const unsigned gN   = (N_NODES + T - 1) / T;
  const unsigned gGemm = (unsigned)((N_NODES / 16 + 7) / 8);                // 391
  const unsigned gW   = (HID * HID + T - 1) / T;

  // degree -> reciprocal (shared by both layers)
  k_zero<<<gN, T, 0, stream>>>(rcnt, N_NODES);
  k_degree<<<gE, T, 0, stream>>>(edst, rcnt, N_EDGES);
  k_rcnt<<<gN, T, 0, stream>>>(rcnt, N_NODES);

  // h0 and bf16 weights
  k_build_h0<<<gNF, T, 0, stream>>>(x, st, emb, h0bf, N_NODES);
  k_f32_to_bf16<<<gW, T, 0, stream>>>(Wl1, wl1bf, HID * HID);
  k_f32_to_bf16<<<gW, T, 0, stream>>>(Wr1, wr1bf, HID * HID);
  k_f32_to_bf16<<<gW, T, 0, stream>>>(Wl2, wl2bf, HID * HID);
  k_f32_to_bf16<<<gW, T, 0, stream>>>(Wr2, wr2bf, HID * HID);

  // ---- layer 1 ----
  k_zero<<<gNF, T, 0, stream>>>(agg, NF);
  k_scatter<<<gEF, T, 0, stream>>>(esrc, edst, h0bf, agg, N_EDGES);
  k_meanbf<<<gNF, T, 0, stream>>>(agg, rcnt, mbf, NF);
  k_gemm_sage<<<gGemm, T, 0, stream>>>(mbf, wl1bf, bl1, h0bf, wr1bf,
                                       nullptr, h1bf, N_NODES);

  // ---- layer 2 ----
  k_zero<<<gNF, T, 0, stream>>>(agg, NF);
  k_scatter<<<gEF, T, 0, stream>>>(esrc, edst, h1bf, agg, N_EDGES);
  k_meanbf<<<gNF, T, 0, stream>>>(agg, rcnt, mbf, NF);
  k_gemm_sage<<<gGemm, T, 0, stream>>>(mbf, wl2bf, bl2, h1bf, wr2bf,
                                       h2f, nullptr, N_NODES);

  // ---- pooling + head ----
  k_zero<<<(NUM_GRAPHS * HID + NUM_GRAPHS + T - 1) / T, T, 0, stream>>>(
      gsum, NUM_GRAPHS * HID + NUM_GRAPHS);
  k_pool<<<gNF, T, 0, stream>>>(h2f, bat, gsum, gcnt, N_NODES);
  k_head<<<(NUM_GRAPHS * OUT_DIM + T - 1) / T, T, 0, stream>>>(
      gsum, gcnt, Wout, bout, out);
}
